// Model_10823317586508
// MI455X (gfx1250) — matrix operand order repaired and checked by reading, not compiled, not run on hardware
//
#include <hip/hip_runtime.h>
#include <hip/hip_bf16.h>
#include <math.h>

// ---------------------------------------------------------------------------
// Attention-LSTM decoder for MI455X (gfx1250, wave32, WMMA).
// All GEMMs run through v_wmma_f32_16x16x32_bf16 (f32 accumulate).
// Each GEMM wave computes a 32x64 tile (2 M-tiles x 4 N-tiles): every B-tile
// load feeds 2 WMMAs, 8 WMMAs per K-slice hide the VMEM waits.
// One-hot @ W_ih is folded into a column gather (K=1256 -> K=512 GEMM).
// ---------------------------------------------------------------------------

typedef __attribute__((ext_vector_type(16))) __bf16 v16bf;
typedef __attribute__((ext_vector_type(8)))  __bf16 v8bf;
typedef __attribute__((ext_vector_type(8)))  float  v8f;

// ---- problem constants ----
#define B_    2048
#define T_    64
#define D_    256
#define H_    256
#define C_    1000
#define STEPS 26
#define WIHLD 1256   // D + C

// ---- workspace layout (bytes, all 256B aligned) ----
#define OFF_WI2H   ((size_t)0)           // [256 x 256] bf16            131072
#define OFF_WH2H   ((size_t)131072)      // [256 x 256] bf16 (attn h2h) 131072
#define OFF_WCAT   ((size_t)262144)      // [1024 x 512] bf16           1048576
#define OFF_WGEN   ((size_t)1310720)     // [1024 x 256] bf16 (padded)  524288
#define OFF_HPROJ  ((size_t)1835008)     // [B*T x 256] bf16            67108864
#define OFF_X      ((size_t)68943872)    // [B x 512] bf16 (ctx||h)     2097152
#define OFF_HP     ((size_t)71041024)    // [B x 256] f32               2097152
#define OFF_C      ((size_t)73138176)    // [B x 256] f32               2097152
#define OFF_GATES  ((size_t)75235328)    // [B x 1024] f32              8388608
#define OFF_HID    ((size_t)83623936)    // [B*26 x 256] bf16           27262976
// total: 110886912 bytes (~106 MB)

// ---------------------------------------------------------------------------
// WMMA helpers (layouts per CDNA5 ISA 7.12.2)
// ---------------------------------------------------------------------------
__device__ __forceinline__ v8f wmma_bf16(v16bf a, v16bf b, v8f c) {
  return __builtin_amdgcn_wmma_f32_16x16x32_bf16(false, a, false, b, (short)0, c,
                                                 false, false);
}

// A tile 16x32 bf16 from row-major [M x K] bf16, leading dim lda.
// lanes 0-15: row=lane,   K = kb+{0..7, 16..23}
// lanes16-31: row=lane-16,K = kb+{8..15,24..31}
__device__ __forceinline__ v16bf load_a_bf16(const __bf16* A, int lda, int mBase,
                                             int kb, int lane) {
  const int m  = mBase + (lane & 15);
  const int ko = (lane >> 4) << 3;
  const __bf16* p = A + (size_t)m * lda + kb + ko;
  v8bf lo = *(const v8bf*)p;
  v8bf hi = *(const v8bf*)(p + 16);
  return __builtin_shufflevector(lo, hi, 0,1,2,3,4,5,6,7,8,9,10,11,12,13,14,15);
}

// Same but converting from f32 source on the fly (v_cvt_pk_bf16_f32).
__device__ __forceinline__ v16bf load_a_f32(const float* A, int lda, int mBase,
                                            int kb, int lane) {
  const int m  = mBase + (lane & 15);
  const int ko = (lane >> 4) << 3;
  const float* p = A + (size_t)m * lda + kb + ko;
  v16bf a;
#pragma unroll
  for (int j = 0; j < 8; ++j) { a[j] = (__bf16)p[j]; a[8 + j] = (__bf16)p[j + 16]; }
  return a;
}

// B tile 32x16 bf16 from weights stored [N x K] row-major (GEMM is X @ W^T).
// lanes 0-15: col=lane,   K=kb+{0..7, 16..23} ; lanes16-31: col=lane-16, K=kb+{8..15,24..31}
// -> 2x b128 loads (matches A layout).
__device__ __forceinline__ v16bf load_b_bf16(const __bf16* W, int ldb, int nBase,
                                             int kb, int lane) {
  const int n  = nBase + (lane & 15);
  const int ko = (lane >> 4) << 3;
  const __bf16* p = W + (size_t)n * ldb + kb + ko;
  v8bf lo = *(const v8bf*)p;
  v8bf hi = *(const v8bf*)(p + 16);
  return __builtin_shufflevector(lo, hi, 0,1,2,3,4,5,6,7,8,9,10,11,12,13,14,15);
}

// ---------------------------------------------------------------------------
// Prep: convert weights to bf16, build Wcat = [W_ih[:, :256] || W_hh],
// pad W_gen to 1024 rows, zero initial state (h part of X, and c).
// ---------------------------------------------------------------------------
__global__ void k_prep(const float* __restrict__ Wi2h_in,
                       const float* __restrict__ Wh2h_in,
                       const float* __restrict__ Wih_in,
                       const float* __restrict__ Whh_in,
                       const float* __restrict__ Wgen_in,
                       __bf16* __restrict__ Wi2h, __bf16* __restrict__ Wh2h,
                       __bf16* __restrict__ Wcat, __bf16* __restrict__ Wgen,
                       __bf16* __restrict__ X, float* __restrict__ c) {
  long i = (long)blockIdx.x * blockDim.x + threadIdx.x;
  if (i < 65536) { Wi2h[i] = (__bf16)Wi2h_in[i]; return; }
  i -= 65536;
  if (i < 65536) { Wh2h[i] = (__bf16)Wh2h_in[i]; return; }
  i -= 65536;
  if (i < 524288) {                       // Wcat [1024 x 512]
    const int n = (int)(i >> 9), k = (int)(i & 511);
    const float v = (k < 256) ? Wih_in[(size_t)n * WIHLD + k]
                              : Whh_in[(size_t)n * 256 + (k - 256)];
    Wcat[i] = (__bf16)v; return;
  }
  i -= 524288;
  if (i < 262144) {                       // Wgen padded [1024 x 256]
    const int n = (int)(i >> 8), k = (int)(i & 255);
    Wgen[i] = (n < C_) ? (__bf16)Wgen_in[(size_t)n * 256 + k] : (__bf16)0.f;
    return;
  }
  i -= 262144;
  if (i < 1048576) { X[i] = (__bf16)0.f; return; }   // ctx||h = 0
  i -= 1048576;
  if (i < 524288) { c[i] = 0.f; return; }            // c = 0
}

// ---------------------------------------------------------------------------
// H_proj = batch_H @ W_i2h^T   [131072 x 256] x [256 x 256] -> bf16
// One wave = 32x64 tile; 4096 M-pairs * 4 N-groups = 16384 waves.
// ---------------------------------------------------------------------------
__global__ void k_i2h(const float* __restrict__ BH, const __bf16* __restrict__ W,
                      __bf16* __restrict__ Hproj) {
  const int lane = threadIdx.x & 31;
  const int w = (blockIdx.x * blockDim.x + threadIdx.x) >> 5;
  const int mBase = (w >> 2) << 5;
  const int nBase = (w & 3) << 6;
  v8f acc[2][4] = {};
  for (int kb = 0; kb < 256; kb += 32) {
    // prefetch next K-slice of the streaming A operand (global_prefetch_b8)
    if (kb + 32 < 256) {
      __builtin_prefetch(BH + (size_t)(mBase + (lane & 15)) * 256 + kb + 32, 0, 1);
      __builtin_prefetch(BH + (size_t)(mBase + 16 + (lane & 15)) * 256 + kb + 32, 0, 1);
    }
    v16bf a0 = load_a_f32(BH, 256, mBase, kb, lane);
    v16bf a1 = load_a_f32(BH, 256, mBase + 16, kb, lane);
#pragma unroll
    for (int t = 0; t < 4; ++t) {
      v16bf b = load_b_bf16(W, 256, nBase + t * 16, kb, lane);
      acc[0][t] = wmma_bf16(a0, b, acc[0][t]);
      acc[1][t] = wmma_bf16(a1, b, acc[1][t]);
    }
  }
  const int n0 = lane & 15;
  const int mrow = mBase + ((lane >> 4) << 3);
#pragma unroll
  for (int mi = 0; mi < 2; ++mi)
#pragma unroll
    for (int r = 0; r < 8; ++r) {
      const size_t ro = (size_t)(mrow + mi * 16 + r) * 256;
#pragma unroll
      for (int t = 0; t < 4; ++t)
        Hproj[ro + nBase + t * 16 + n0] = (__bf16)acc[mi][t][r];
    }
}

// ---------------------------------------------------------------------------
// hp = h @ W_h2h^T + b_h2h   [2048 x 256] x [256 x 256] -> f32 (256 waves)
// ---------------------------------------------------------------------------
__global__ void k_hp(const __bf16* __restrict__ X, const __bf16* __restrict__ W,
                     const float* __restrict__ bias, float* __restrict__ hp) {
  const int lane = threadIdx.x & 31;
  const int w = (blockIdx.x * blockDim.x + threadIdx.x) >> 5;
  const int mBase = (w >> 2) << 5;
  const int nBase = (w & 3) << 6;
  const __bf16* A = X + 256;   // h lives in X[:, 256:512]
  v8f acc[2][4] = {};
  for (int kb = 0; kb < 256; kb += 32) {
    v16bf a0 = load_a_bf16(A, 512, mBase, kb, lane);
    v16bf a1 = load_a_bf16(A, 512, mBase + 16, kb, lane);
#pragma unroll
    for (int t = 0; t < 4; ++t) {
      v16bf b = load_b_bf16(W, 256, nBase + t * 16, kb, lane);
      acc[0][t] = wmma_bf16(a0, b, acc[0][t]);
      acc[1][t] = wmma_bf16(a1, b, acc[1][t]);
    }
  }
  const int n0 = lane & 15;
  const int mrow = mBase + ((lane >> 4) << 3);
#pragma unroll
  for (int mi = 0; mi < 2; ++mi)
#pragma unroll
    for (int r = 0; r < 8; ++r) {
      const size_t ro = (size_t)(mrow + mi * 16 + r) * 256;
#pragma unroll
      for (int t = 0; t < 4; ++t) {
        const int col = nBase + t * 16 + n0;
        hp[ro + col] = acc[mi][t][r] + bias[col];
      }
    }
}

// ---------------------------------------------------------------------------
// Fused attention: e = Wscore . tanh(Hproj + hp); softmax over T;
// ctx = alpha @ batch_H. One 256-thread block per batch row.
// ---------------------------------------------------------------------------
__global__ void k_attn(const __bf16* __restrict__ Hproj, const float* __restrict__ hp,
                       const float* __restrict__ Wscore, const float* __restrict__ BH,
                       __bf16* __restrict__ X) {
  __shared__ float s_hp[256], s_ws[256], s_part[256], s_e[64], s_alpha[64];
  const int b = blockIdx.x;
  const int tid = threadIdx.x;
  s_hp[tid] = hp[(size_t)b * 256 + tid];
  s_ws[tid] = Wscore[tid];
  __syncthreads();

  // scores: 4 threads per timestep, 64 H-elements each
  const int t  = tid >> 2;
  const int h0 = (tid & 3) << 6;
  const __bf16* Hp = Hproj + ((size_t)b * T_ + t) * 256 + h0;
  const float* hpl = s_hp + h0;
  const float* wsl = s_ws + h0;
  float partial = 0.f;
#pragma unroll 8
  for (int j = 0; j < 64; ++j)
    partial += wsl[j] * tanhf((float)Hp[j] + hpl[j]);
  s_part[tid] = partial;
  __syncthreads();
  if (tid < 64)
    s_e[tid] = s_part[4 * tid] + s_part[4 * tid + 1] +
               s_part[4 * tid + 2] + s_part[4 * tid + 3];
  __syncthreads();

  // softmax over T=64 (broadcast reads from LDS, redundant but cheap)
  float mx = -1e30f;
  for (int j = 0; j < 64; ++j) mx = fmaxf(mx, s_e[j]);
  float sum = 0.f;
  for (int j = 0; j < 64; ++j) sum += __expf(s_e[j] - mx);
  if (tid < 64) s_alpha[tid] = __expf(s_e[tid] - mx) / sum;
  __syncthreads();

  // context: thread tid owns feature d=tid
  const float* bh = BH + (size_t)b * T_ * 256 + tid;
  float ctx = 0.f;
#pragma unroll 8
  for (int tt = 0; tt < 64; ++tt) ctx += s_alpha[tt] * bh[tt * 256];
  X[(size_t)b * 512 + tid] = (__bf16)ctx;
}

// ---------------------------------------------------------------------------
// gates = [ctx||h] @ [W_ih[:, :256] || W_hh]^T + b_ih + b_hh + W_ih[:, 256+char]
// [2048 x 512] x [512 x 1024] -> f32, one-hot folded into gather epilogue.
// 64 M-pairs * 16 N-groups = 1024 waves.
// ---------------------------------------------------------------------------
__global__ void k_gates(const __bf16* __restrict__ X, const __bf16* __restrict__ Wcat,
                        const float* __restrict__ b_ih, const float* __restrict__ b_hh,
                        const float* __restrict__ W_ih, const int* __restrict__ text,
                        int step, float* __restrict__ gates) {
  const int lane = threadIdx.x & 31;
  const int w = (blockIdx.x * blockDim.x + threadIdx.x) >> 5;
  const int mBase = (w >> 4) << 5;
  const int nBase = (w & 15) << 6;
  v8f acc[2][4] = {};
  for (int kb = 0; kb < 512; kb += 32) {
    v16bf a0 = load_a_bf16(X, 512, mBase, kb, lane);
    v16bf a1 = load_a_bf16(X, 512, mBase + 16, kb, lane);
#pragma unroll
    for (int t = 0; t < 4; ++t) {
      v16bf b = load_b_bf16(Wcat, 512, nBase + t * 16, kb, lane);
      acc[0][t] = wmma_bf16(a0, b, acc[0][t]);
      acc[1][t] = wmma_bf16(a1, b, acc[1][t]);
    }
  }
  const int n0 = lane & 15;
  const int mrow = mBase + ((lane >> 4) << 3);
#pragma unroll
  for (int mi = 0; mi < 2; ++mi)
#pragma unroll
    for (int r = 0; r < 8; ++r) {
      const int row = mrow + mi * 16 + r;
      const int ch = text[row * STEPS + step];
      const size_t ro = (size_t)row * 1024;
#pragma unroll
      for (int t = 0; t < 4; ++t) {
        const int col = nBase + t * 16 + n0;
        gates[ro + col] = acc[mi][t][r] + b_ih[col] + b_hh[col] +
                          W_ih[(size_t)col * WIHLD + D_ + ch];
      }
    }
}

// ---------------------------------------------------------------------------
// LSTM pointwise; writes h (bf16) into next-step X and into hiddens buffer.
// ---------------------------------------------------------------------------
__device__ __forceinline__ float sigmoidf_(float x) { return 1.f / (1.f + __expf(-x)); }

__global__ void k_lstm(const float* __restrict__ gates, float* __restrict__ c,
                       __bf16* __restrict__ X, __bf16* __restrict__ hid, int step) {
  const int idx = blockIdx.x * blockDim.x + threadIdx.x;  // B*H
  const int b = idx >> 8;
  const int j = idx & 255;
  const size_t g = (size_t)b * 1024 + j;
  const float ig = gates[g], fg = gates[g + 256], gg = gates[g + 512], og = gates[g + 768];
  const float cn = sigmoidf_(fg) * c[idx] + sigmoidf_(ig) * tanhf(gg);
  const float hn = sigmoidf_(og) * tanhf(cn);
  c[idx] = cn;
  const __bf16 hb = (__bf16)hn;
  X[(size_t)b * 512 + 256 + j] = hb;
  hid[((size_t)b * STEPS + step) * 256 + j] = hb;
}

// ---------------------------------------------------------------------------
// probs = hiddens @ W_gen^T + b_gen   [53248 x 256] x [256 x 1000(pad 1024)]
// 1664 M-pairs * 16 N-groups = 26624 waves.
// ---------------------------------------------------------------------------
__global__ void k_gen(const __bf16* __restrict__ hid, const __bf16* __restrict__ Wgen,
                      const float* __restrict__ b_gen, float* __restrict__ out) {
  const int lane = threadIdx.x & 31;
  const int w = (blockIdx.x * blockDim.x + threadIdx.x) >> 5;
  const int mBase = (w >> 4) << 5;
  const int nBase = (w & 15) << 6;
  v8f acc[2][4] = {};
  for (int kb = 0; kb < 256; kb += 32) {
    if (kb + 32 < 256) {
      __builtin_prefetch(hid + (size_t)(mBase + (lane & 15)) * 256 + kb + 32, 0, 1);
      __builtin_prefetch(hid + (size_t)(mBase + 16 + (lane & 15)) * 256 + kb + 32, 0, 1);
    }
    v16bf a0 = load_a_bf16(hid, 256, mBase, kb, lane);
    v16bf a1 = load_a_bf16(hid, 256, mBase + 16, kb, lane);
#pragma unroll
    for (int t = 0; t < 4; ++t) {
      v16bf b = load_b_bf16(Wgen, 256, nBase + t * 16, kb, lane);
      acc[0][t] = wmma_bf16(a0, b, acc[0][t]);
      acc[1][t] = wmma_bf16(a1, b, acc[1][t]);
    }
  }
  const int n0 = lane & 15;
  const int mrow = mBase + ((lane >> 4) << 3);
#pragma unroll
  for (int mi = 0; mi < 2; ++mi)
#pragma unroll
    for (int r = 0; r < 8; ++r) {
      const size_t ro = (size_t)(mrow + mi * 16 + r) * C_;
#pragma unroll
      for (int t = 0; t < 4; ++t) {
        const int col = nBase + t * 16 + n0;
        if (col < C_) out[ro + col] = acc[mi][t][r] + b_gen[col];
      }
    }
}

// ---------------------------------------------------------------------------
extern "C" void kernel_launch(void* const* d_in, const int* in_sizes, int n_in,
                              void* d_out, int out_size, void* d_ws, size_t ws_size,
                              hipStream_t stream) {
  const float* batch_H = (const float*)d_in[0];
  const int*   text    = (const int*)d_in[1];
  const float* W_i2h   = (const float*)d_in[2];
  const float* W_h2h   = (const float*)d_in[3];
  const float* b_h2h   = (const float*)d_in[4];
  const float* W_score = (const float*)d_in[5];
  const float* W_ih    = (const float*)d_in[6];
  const float* W_hh    = (const float*)d_in[7];
  const float* b_ih    = (const float*)d_in[8];
  const float* b_hh    = (const float*)d_in[9];
  const float* W_gen   = (const float*)d_in[10];
  const float* b_gen   = (const float*)d_in[11];
  float* out = (float*)d_out;

  char* ws = (char*)d_ws;
  __bf16* Wi2h_b = (__bf16*)(ws + OFF_WI2H);
  __bf16* Wh2h_b = (__bf16*)(ws + OFF_WH2H);
  __bf16* Wcat_b = (__bf16*)(ws + OFF_WCAT);
  __bf16* Wgen_b = (__bf16*)(ws + OFF_WGEN);
  __bf16* Hproj  = (__bf16*)(ws + OFF_HPROJ);
  __bf16* X      = (__bf16*)(ws + OFF_X);
  float*  hp     = (float*)(ws + OFF_HP);
  float*  c      = (float*)(ws + OFF_C);
  float*  gates  = (float*)(ws + OFF_GATES);
  __bf16* hid    = (__bf16*)(ws + OFF_HID);

  // weights -> bf16, state -> 0  (2,490,368 work items)
  k_prep<<<9728, 256, 0, stream>>>(W_i2h, W_h2h, W_ih, W_hh, W_gen,
                                   Wi2h_b, Wh2h_b, Wcat_b, Wgen_b, X, c);

  // step-invariant i2h projection (16384 waves, 32x64 tile per wave)
  k_i2h<<<2048, 256, 0, stream>>>(batch_H, Wi2h_b, Hproj);

  for (int s = 0; s < STEPS; ++s) {
    k_hp   <<<32,   256, 0, stream>>>(X, Wh2h_b, b_h2h, hp);
    k_attn <<<B_,   256, 0, stream>>>(Hproj, hp, W_score, batch_H, X);
    k_gates<<<128,  256, 0, stream>>>(X, Wcat_b, b_ih, b_hh, W_ih, text, s, gates);
    k_lstm <<<2048, 256, 0, stream>>>(gates, c, X, hid, s);
  }

  // final classifier (26624 waves, 32x64 tile per wave)
  k_gen<<<3328, 256, 0, stream>>>(hid, Wgen_b, b_gen, out);
}
